// SelfAttention_65481071409462
// MI455X (gfx1250) — matrix operand order repaired and checked by reading, not compiled, not run on hardware
//
#include <hip/hip_runtime.h>
#include <hip/hip_bf16.h>
#include <math.h>

typedef __attribute__((ext_vector_type(16))) _Float16 v16h;
typedef __attribute__((ext_vector_type(8)))  _Float16 v8h;
typedef __attribute__((ext_vector_type(8)))  float    v8f;
typedef __attribute__((ext_vector_type(4)))  float    v4f;

// ---------------------------------------------------------------------------
// WMMA wrapper: D = A(16x32 f16) * B(32x16 f16) + C(16x16 f32)
// ---------------------------------------------------------------------------
__device__ __forceinline__ v8f wmma16(v16h a, v16h b, v8f c) {
  return __builtin_amdgcn_wmma_f32_16x16x32_f16(
      /*neg_a=*/false, a, /*neg_b=*/false, b,
      /*c_mod=*/(short)0, c, /*reuse_a=*/false, /*reuse_b=*/false);
}

// ---------------------------------------------------------------------------
// A-fragment (16x32, M x K) from f32 row-major [row*ld + k].
// Lane r=lane&15 holds row r. kb = (lane>=16) ? 8 : 0.
// halves 0..7  = K kb..kb+7 ;  halves 8..15 = K kb+16..kb+23.
// ---------------------------------------------------------------------------
__device__ __forceinline__ v16h load_a_f32(const float* __restrict__ A,
                                           int row, int k0, int ld, int lane) {
  int kb = (lane & 16) ? 8 : 0;
  const float* p = A + (size_t)row * ld + k0 + kb;
  v4f f0 = *(const v4f*)(p);
  v4f f1 = *(const v4f*)(p + 4);
  v4f f2 = *(const v4f*)(p + 16);
  v4f f3 = *(const v4f*)(p + 20);
  v16h a;
#pragma unroll
  for (int t = 0; t < 4; ++t) {
    a[t]      = (_Float16)f0[t];
    a[4 + t]  = (_Float16)f1[t];
    a[8 + t]  = (_Float16)f2[t];
    a[12 + t] = (_Float16)f3[t];
  }
  return a;
}

// Same A-fragment layout from f16 row-major source (global or LDS).
__device__ __forceinline__ v16h load_a_f16(const _Float16* A,
                                           int row, int k0, int ld, int lane) {
  int kb = (lane & 16) ? 8 : 0;
  const _Float16* p = A + (size_t)row * ld + k0 + kb;
  v8h lo = *(const v8h*)(p);        // K kb..kb+7
  v8h hi = *(const v8h*)(p + 16);   // K kb+16..kb+23
  v16h a;
#pragma unroll
  for (int t = 0; t < 8; ++t) { a[t] = lo[t]; a[8 + t] = hi[t]; }
  return a;
}

// ---------------------------------------------------------------------------
// B-fragment (32x16, K x N) from "Bt": f16 storage where output column n is a
// contiguous run of K values (Bt[n*ld + k]). Lane n=lane&15 holds column n.
// kb = (lane>=16) ? 8 : 0 ; halves 0..7 = K kb..kb+7 ; halves 8..15 = K kb+16..kb+23.
// ---------------------------------------------------------------------------
__device__ __forceinline__ v16h load_b_f16(const _Float16* Bt,
                                           int nrow, int k0, int ld, int lane) {
  int kb = (lane & 16) ? 8 : 0;
  const _Float16* p = Bt + (size_t)nrow * ld + k0 + kb;
  v8h lo = *(const v8h*)(p);
  v8h hi = *(const v8h*)(p + 16);
  v16h b;
#pragma unroll
  for (int t = 0; t < 8; ++t) { b[t] = lo[t]; b[8 + t] = hi[t]; }
  return b;
}

// LDS byte offset of a generic pointer derived from a __shared__ array:
// the low 32 bits of the generic (flat) LDS address are the LDS offset.
__device__ __forceinline__ unsigned lds_off(const void* p) {
  return (unsigned)(size_t)p;
}

// ---------------------------------------------------------------------------
// Weight prep: f32 (K x N) row-major  ->  f16 transposed (N x K)
// ---------------------------------------------------------------------------
__global__ void transpose_cvt_kernel(const float* __restrict__ W,
                                     _Float16* __restrict__ Wt, int K, int N) {
  int i = blockIdx.x * 256 + threadIdx.x;
  if (i >= K * N) return;
  int k = i / N, n = i % N;
  Wt[(size_t)n * K + k] = (_Float16)W[i];
}

// ---------------------------------------------------------------------------
// P = relu(X @ proj_W + b)   X:(4096x1024 f32)  pWt:(512x1024 f16)  P:(4096x512 f16)
// One wave -> 16x64 tile. K-loop fully unrolled: no loop-carried registers
// except the accumulators, so the scheduler hoists loads across steps with
// no rotation moves / WAR hazard NOPs.
// ---------------------------------------------------------------------------
__global__ void __launch_bounds__(256)
gemm_proj_kernel(const float* __restrict__ X, const _Float16* __restrict__ pWt,
                 const float* __restrict__ bias, _Float16* __restrict__ P) {
  int lane = threadIdx.x & 31, wave = threadIdx.x >> 5;
  int w  = blockIdx.x * 8 + wave;   // 2048 waves
  int tn = w & 7;                   // 512/64 = 8 column tiles
  int tm = w >> 3;                  // 4096/16 = 256 row tiles
  int row = tm * 16 + (lane & 15);
  int n0  = tn * 64;
  v8f c[4] = {};
#pragma unroll
  for (int k0 = 0; k0 < 1024; k0 += 32) {
    if ((k0 & 96) == 0 && k0 + 128 < 1024)
      __builtin_prefetch(X + (size_t)row * 1024 + k0 + 128, 0, 0);
    v16h a = load_a_f32(X, row, k0, 1024, lane);
    v16h b[4];
#pragma unroll
    for (int j = 0; j < 4; ++j)
      b[j] = load_b_f16(pWt, n0 + j * 16 + (lane & 15), k0, 1024, lane);
#pragma unroll
    for (int j = 0; j < 4; ++j) c[j] = wmma16(a, b[j], c[j]);
  }
  int n = lane & 15, g = lane >> 4;
#pragma unroll
  for (int j = 0; j < 4; ++j)
#pragma unroll
    for (int i = 0; i < 8; ++i) {
      int r   = tm * 16 + i + 8 * g;
      int col = n0 + j * 16 + n;
      float v = c[j][i] + bias[col];
      v = v > 0.0f ? v : 0.0f;
      P[(size_t)r * 512 + col] = (_Float16)v;
    }
}

// ---------------------------------------------------------------------------
// Q/K/V = P @ W  (M=4096, N=512, K=512). blockIdx.y selects matrix.
// Q,K stored (bh, l, d) row-major; V stored transposed (bh, d, l).
// ---------------------------------------------------------------------------
__global__ void __launch_bounds__(256)
gemm_qkv_kernel(const _Float16* __restrict__ P,
                const _Float16* __restrict__ qWt, const _Float16* __restrict__ kWt,
                const _Float16* __restrict__ vWt,
                _Float16* __restrict__ Q, _Float16* __restrict__ K,
                _Float16* __restrict__ Vt) {
  int sel = blockIdx.y;
  const _Float16* Wt = (sel == 0) ? qWt : ((sel == 1) ? kWt : vWt);
  int lane = threadIdx.x & 31, wave = threadIdx.x >> 5;
  int w  = blockIdx.x * 8 + wave;
  int tn = w & 7;
  int tm = w >> 3;
  int row = tm * 16 + (lane & 15);
  int n0  = tn * 64;
  v8f c[4] = {};
#pragma unroll
  for (int k0 = 0; k0 < 512; k0 += 32) {
    v16h a = load_a_f16(P, row, k0, 512, lane);
    v16h b[4];
#pragma unroll
    for (int j = 0; j < 4; ++j)
      b[j] = load_b_f16(Wt, n0 + j * 16 + (lane & 15), k0, 512, lane);
#pragma unroll
    for (int j = 0; j < 4; ++j) c[j] = wmma16(a, b[j], c[j]);
  }
  int n = lane & 15, g = lane >> 4;
#pragma unroll
  for (int j = 0; j < 4; ++j)
#pragma unroll
    for (int i = 0; i < 8; ++i) {
      int r   = tm * 16 + i + 8 * g;       // global token 0..4095
      int col = n0 + j * 16 + n;           // 0..511 = h*64+d
      int b_  = r >> 11, lq = r & 2047;
      int h   = col >> 6, d  = col & 63;
      _Float16 v = (_Float16)c[j][i];
      size_t bh = (size_t)(b_ * 8 + h);
      if (sel == 0)      Q[(bh * 2048 + lq) * 64 + d]  = v;
      else if (sel == 1) K[(bh * 2048 + lq) * 64 + d]  = v;
      else               Vt[(bh * 64 + d) * 2048 + lq] = v;
    }
}

// ---------------------------------------------------------------------------
// Flash attention. All 8 waves of a block share one (b,h): the 32-key K tile
// (32x64 f16) and V tile (64x32 f16, d-major) are staged into LDS ONCE per
// block with global_load_async_to_lds_b128 (ASYNCcnt), then every wave reads
// its WMMA fragments from LDS. Online softmax in registers; the P tile is
// bounced through a per-wave 1KB LDS region to reshape into an A-fragment.
// grid.x = 16 bh * 16 qblocks = 256 ; block = 256 (8 waves, 128 q rows/block)
// ---------------------------------------------------------------------------
__global__ void __launch_bounds__(256)
fa_kernel(const _Float16* __restrict__ Q, const _Float16* __restrict__ K,
          const _Float16* __restrict__ Vt, _Float16* __restrict__ heads) {
  __shared__ _Float16 sK[32 * 64];   // K tile: 32 keys x 64 d      (4 KB)
  __shared__ _Float16 sV[64 * 32];   // V tile: 64 d    x 32 keys   (4 KB)
  __shared__ _Float16 sP[8 * 512];   // per-wave P tiles 16x32      (8 KB)

  int lane = threadIdx.x & 31, wave = threadIdx.x >> 5;
  int t = threadIdx.x;
  _Float16* sw = sP + wave * 512;

  int bh = blockIdx.x >> 4;
  int qb = blockIdx.x & 15;
  int b_ = bh >> 3, h = bh & 7;
  const _Float16* Qh = Q  + (size_t)bh * 2048 * 64;
  const _Float16* Kh = K  + (size_t)bh * 2048 * 64;
  const _Float16* Vh = Vt + (size_t)bh * 64 * 2048;
  int qrow = qb * 128 + wave * 16;

  v16h aq0 = load_a_f16(Qh, qrow + (lane & 15), 0, 64, lane);
  v16h aq1 = load_a_f16(Qh, qrow + (lane & 15), 32, 64, lane);

  // per-thread async-copy slots (loop-invariant parts)
  int kr = t >> 3, kc = (t & 7) * 8;        // K tile: 32 rows x 8 chunks
  int vr = t >> 2, vc = (t & 3) * 8;        // V tile: 64 rows x 4 chunks
  unsigned lK = lds_off(sK + kr * 64 + kc);
  unsigned lV = lds_off(sV + vr * 32 + vc);

  float m[8], l[8];
  v8f o[4] = {};
#pragma unroll
  for (int i = 0; i < 8; ++i) { m[i] = -INFINITY; l[i] = 0.0f; }
  const float scale = 0.125f;   // 1/sqrt(64)
  int n = lane & 15, g = lane >> 4;

  for (int kt = 0; kt < 2048; kt += 32) {
    // ---- stage K/V tiles into LDS via async DMA (one 16B chunk x2 / thread)
    const _Float16* gK = Kh + (size_t)(kt + kr) * 64 + kc;
    const _Float16* gV = Vh + (size_t)vr * 2048 + kt + vc;
    asm volatile("global_load_async_to_lds_b128 %0, %1, off"
                 :: "v"(lK), "v"(gK) : "memory");
    asm volatile("global_load_async_to_lds_b128 %0, %1, off"
                 :: "v"(lV), "v"(gV) : "memory");
    asm volatile("s_wait_asynccnt 0x0" ::: "memory");
    __syncthreads();                       // tiles visible to all waves

    // ---- S tile: 16 q x 32 k (two 16x16 accumulators), K frags from LDS
    v8f s0 = {}, s1 = {};
    v16h bk0 = load_b_f16(sK, n, 0, 64, lane);
    v16h bk1 = load_b_f16(sK, n, 32, 64, lane);
    v16h bk2 = load_b_f16(sK, 16 + n, 0, 64, lane);
    v16h bk3 = load_b_f16(sK, 16 + n, 32, 64, lane);
    s0 = wmma16(aq0, bk0, s0);
    s0 = wmma16(aq1, bk1, s0);
    s1 = wmma16(aq0, bk2, s1);
    s1 = wmma16(aq1, bk3, s1);

    // ---- online softmax per C-layout row (row = i+8g, cols = lanes 0..15)
#pragma unroll
    for (int i = 0; i < 8; ++i) {
      float x0 = s0[i] * scale, x1 = s1[i] * scale;
      float mx = fmaxf(x0, x1);
#pragma unroll
      for (int msk = 1; msk < 16; msk <<= 1)
        mx = fmaxf(mx, __shfl_xor(mx, msk, 32));
      float mn    = fmaxf(m[i], mx);
      float alpha = __expf(m[i] - mn);
      float p0 = __expf(x0 - mn), p1 = __expf(x1 - mn);
      float rs = p0 + p1;
#pragma unroll
      for (int msk = 1; msk < 16; msk <<= 1)
        rs += __shfl_xor(rs, msk, 32);
      l[i] = l[i] * alpha + rs;
      m[i] = mn;
#pragma unroll
      for (int j = 0; j < 4; ++j) o[j][i] *= alpha;
      int row = i + 8 * g;
      sw[row * 32 + n]      = (_Float16)p0;
      sw[row * 32 + 16 + n] = (_Float16)p1;
    }
    __syncthreads();                       // order P writes vs fragment reads

    // ---- O += P (16x32) @ V (32x64), V frags from LDS (d-major)
    int kbp = (lane & 16) ? 8 : 0;
    const _Float16* pp = sw + (lane & 15) * 32 + kbp;
    v8h plo = *(const v8h*)(pp);
    v8h phi = *(const v8h*)(pp + 16);
    v16h ap;
#pragma unroll
    for (int tt = 0; tt < 8; ++tt) { ap[tt] = plo[tt]; ap[8 + tt] = phi[tt]; }
#pragma unroll
    for (int j = 0; j < 4; ++j) {
      v16h bv = load_b_f16(sV, j * 16 + n, 0, 32, lane);
      o[j] = wmma16(ap, bv, o[j]);
    }
    __syncthreads();                       // all reads done before next stage
  }

  // ---- O /= l ; store heads (B, L, H*Dh) as f16
#pragma unroll
  for (int i = 0; i < 8; ++i) {
    int lq = qrow + i + 8 * g;
    float inv = 1.0f / l[i];
#pragma unroll
    for (int j = 0; j < 4; ++j) {
      heads[((size_t)(b_ * 2048 + lq)) * 512 + h * 64 + j * 16 + n] =
          (_Float16)(o[j][i] * inv);
    }
  }
}

// ---------------------------------------------------------------------------
// out = heads @ out_W   (M=4096, N=1024, K=512) -> f32
// ---------------------------------------------------------------------------
__global__ void __launch_bounds__(256)
gemm_out_kernel(const _Float16* __restrict__ H, const _Float16* __restrict__ oWt,
                float* __restrict__ out) {
  int lane = threadIdx.x & 31, wave = threadIdx.x >> 5;
  int w  = blockIdx.x * 8 + wave;   // 4096 waves
  int tn = w & 15;                  // 1024/64 = 16
  int tm = w >> 4;                  // 256
  int row = tm * 16 + (lane & 15);
  int n0  = tn * 64;
  v8f c[4] = {};
#pragma unroll
  for (int k0 = 0; k0 < 512; k0 += 32) {
    v16h a = load_a_f16(H, row, k0, 512, lane);
    v16h b[4];
#pragma unroll
    for (int j = 0; j < 4; ++j)
      b[j] = load_b_f16(oWt, n0 + j * 16 + (lane & 15), k0, 512, lane);
#pragma unroll
    for (int j = 0; j < 4; ++j) c[j] = wmma16(a, b[j], c[j]);
  }
  int n = lane & 15, g = lane >> 4;
#pragma unroll
  for (int j = 0; j < 4; ++j)
#pragma unroll
    for (int i = 0; i < 8; ++i) {
      int r   = tm * 16 + i + 8 * g;
      int col = n0 + j * 16 + n;
      out[(size_t)r * 1024 + col] = c[j][i];
    }
}

// ---------------------------------------------------------------------------
extern "C" void kernel_launch(void* const* d_in, const int* in_sizes, int n_in,
                              void* d_out, int out_size, void* d_ws, size_t ws_size,
                              hipStream_t stream) {
  const float* x      = (const float*)d_in[0];   // (2,2048,1024)
  const float* proj_W = (const float*)d_in[1];   // (1024,512)
  const float* proj_b = (const float*)d_in[2];   // (512,)
  const float* Wq     = (const float*)d_in[3];   // (512,8,64) == (512,512)
  const float* Wk     = (const float*)d_in[4];
  const float* Wv     = (const float*)d_in[5];
  const float* out_W  = (const float*)d_in[6];   // (512,1024)
  float* out = (float*)d_out;

  _Float16* ws16  = (_Float16*)d_ws;
  _Float16* pWt   = ws16;               // 512*1024
  _Float16* qWt   = pWt + 524288;       // 512*512
  _Float16* kWt   = qWt + 262144;
  _Float16* vWt   = kWt + 262144;
  _Float16* oWt   = vWt + 262144;       // 1024*512
  _Float16* P     = oWt + 524288;       // 4096*512
  _Float16* Qm    = P   + 2097152;      // 16*2048*64
  _Float16* Km    = Qm  + 2097152;
  _Float16* Vt    = Km  + 2097152;
  _Float16* heads = Vt  + 2097152;      // 4096*512

  // weight transpose+convert
  transpose_cvt_kernel<<<(1024 * 512 + 255) / 256, 256, 0, stream>>>(proj_W, pWt, 1024, 512);
  transpose_cvt_kernel<<<(512 * 512 + 255) / 256, 256, 0, stream>>>(Wq, qWt, 512, 512);
  transpose_cvt_kernel<<<(512 * 512 + 255) / 256, 256, 0, stream>>>(Wk, kWt, 512, 512);
  transpose_cvt_kernel<<<(512 * 512 + 255) / 256, 256, 0, stream>>>(Wv, vWt, 512, 512);
  transpose_cvt_kernel<<<(512 * 1024 + 255) / 256, 256, 0, stream>>>(out_W, oWt, 512, 1024);

  gemm_proj_kernel<<<256, 256, 0, stream>>>(x, pWt, proj_b, P);
  gemm_qkv_kernel<<<dim3(256, 3), 256, 0, stream>>>(P, qWt, kWt, vWt, Qm, Km, Vt);
  fa_kernel<<<256, 256, 0, stream>>>(Qm, Km, Vt, heads);
  gemm_out_kernel<<<512, 256, 0, stream>>>(heads, oWt, out);
}
